// _SelfAttention_6760278524258
// MI455X (gfx1250) — matrix operand order repaired and checked by reading, not compiled, not run on hardware
//
#include <hip/hip_runtime.h>

#define S_TOT 4096
#define C_DIM 256
#define MID   32
#define NB    4

typedef __attribute__((ext_vector_type(16))) __bf16 v16bf;
typedef __attribute__((ext_vector_type(8)))  float  v8f;

// sched_group_barrier masks: 0x008 = MFMA/WMMA, 0x020 = VMEM read, 0x100 = DS read
#define SGB(mask, size) __builtin_amdgcn_sched_group_barrier((mask), (size), 0)

union FragU { uint4 q[2]; v16bf v; };

// A-matrix 16x32 bf16 fragment: lane holds row M = lane&15,
// K = {0..7,16..23} (lanes 0-15) or {8..15,24..31} (lanes 16-31)
__device__ __forceinline__ v16bf frag_a(const __bf16* rowbase, int lane) {
  const int h = (lane < 16) ? 0 : 8;
  FragU u;
  u.q[0] = *reinterpret_cast<const uint4*>(rowbase + h);
  u.q[1] = *reinterpret_cast<const uint4*>(rowbase + h + 16);
  return u.v;
}

// B-matrix 32x16 bf16 fragment (column-major per lane): lane holds col N = lane&15,
// K = {0..7,16..23} (lanes 0-15) or {8..15,24..31} (lanes 16-31)
__device__ __forceinline__ v16bf frag_b(const __bf16* rowbase, int lane) {
  const int g = (lane < 16) ? 0 : 8;
  FragU u;
  u.q[0] = *reinterpret_cast<const uint4*>(rowbase + g);
  u.q[1] = *reinterpret_cast<const uint4*>(rowbase + g + 16);
  return u.v;
}

// CDNA5 async global->LDS (GVS mode): vdst = LDS byte offset, vaddr = u32 byte
// offset, saddr = 64-bit base. Tracked by ASYNCcnt.
__device__ __forceinline__ void async_load_b128(unsigned lds_off, unsigned goff,
                                                unsigned long long base) {
  asm volatile("global_load_async_to_lds_b128 %0, %1, %2"
               :: "v"(lds_off), "v"(goff), "s"(base)
               : "memory");
}
__device__ __forceinline__ void wait_asynccnt0() {
  asm volatile("s_wait_asynccnt 0x0" ::: "memory");
}

// ---------------- kernel 1: pack Wq/Wk/Wv -> bf16 [320,256], biases -> f32[320]
__global__ void pack_weights(const float* __restrict__ Wq, const float* __restrict__ bq,
                             const float* __restrict__ Wk, const float* __restrict__ bk,
                             const float* __restrict__ Wv, const float* __restrict__ bv,
                             __bf16* __restrict__ Wall, float* __restrict__ ball) {
  int r = blockIdx.x;          // 0..319
  int t = threadIdx.x;         // 0..255
  const float* src;
  float bias;
  if (r < 32)      { src = Wq + (size_t)r * C_DIM;        bias = bq[r];      }
  else if (r < 64) { src = Wk + (size_t)(r - 32) * C_DIM; bias = bk[r - 32]; }
  else             { src = Wv + (size_t)(r - 64) * C_DIM; bias = bv[r - 64]; }
  Wall[(size_t)r * C_DIM + t] = (__bf16)src[t];
  if (t == 0) ball[r] = bias;
}

// ---------------- kernel 2: x [N,C,S] f32 -> xbT [N,S,C] bf16 (LDS tile transpose)
__global__ void transpose_x(const float* __restrict__ x, __bf16* __restrict__ xbT) {
  __shared__ __bf16 tile[64][72];
  const int st = blockIdx.x, ct = blockIdx.y, n = blockIdx.z;
  const int tid = threadIdx.x;
  const int sl = tid & 63, cq = tid >> 6;
  const float* xp = x + (((size_t)n * C_DIM + ct * 64) * S_TOT) + st * 64;
#pragma unroll
  for (int cc = 0; cc < 16; ++cc) {
    int c = cc * 4 + cq;
    tile[c][sl] = (__bf16)xp[(size_t)c * S_TOT + sl];
  }
  __syncthreads();
  const int cl = tid & 63, sq = tid >> 6;
  __bf16* op = xbT + (((size_t)n * S_TOT + st * 64) * C_DIM) + ct * 64;
#pragma unroll
  for (int ss = 0; ss < 16; ++ss) {
    int s = ss * 4 + sq;
    op[(size_t)s * C_DIM + cl] = tile[cl][s];
  }
}

// ---------------- kernel 3: QKV GEMM via WMMA, 512 threads = 16 waves.
// Each wave: 10 M-tiles x one 16-s tile -> acc is only 80 VGPRs, leaving the
// scheduler room to keep fragment loads in flight.
__global__ void __launch_bounds__(512)
qkv_gemm(const __bf16* __restrict__ xbT, const __bf16* __restrict__ Wall,
         const float* __restrict__ ball,
         __bf16* __restrict__ Qt, __bf16* __restrict__ Kt, __bf16* __restrict__ Vv) {
  __shared__ float lds_bias[320];
  const int tid = threadIdx.x;
  for (int i = tid; i < 320; i += 512) lds_bias[i] = ball[i];
  __syncthreads();

  const int w = tid >> 5, lane = tid & 31, l15 = lane & 15;
  const int n = blockIdx.y;
  const int sbase = blockIdx.x * 128 + (w & 7) * 16;
  const int mbase = (w >> 3) * 10;   // M-tile group: 0..9 or 10..19
  const __bf16* xrow = xbT + ((size_t)n * S_TOT + sbase + l15) * C_DIM;
  const __bf16* wrow = Wall + (size_t)(mbase * 16 + l15) * C_DIM;

  v8f acc[10];
#pragma unroll
  for (int mt = 0; mt < 10; ++mt) acc[mt] = (v8f){0, 0, 0, 0, 0, 0, 0, 0};

#pragma unroll 1
  for (int kt = 0; kt < 8; ++kt) {
    v16bf bf = frag_b(xrow + kt * 32, lane);
    v16bf afb[5];
#pragma unroll
    for (int i = 0; i < 5; ++i)
      afb[i] = frag_a(wrow + (size_t)i * 16 * C_DIM + kt * 32, lane);
#pragma unroll
    for (int grp = 0; grp < 2; ++grp) {
#pragma unroll
      for (int i = 0; i < 5; ++i) {
        const int mt = grp * 5 + i;
        v8f r = __builtin_amdgcn_wmma_f32_16x16x32_bf16(false, afb[i], false, bf,
                                                        (short)0, acc[mt], false, false);
        if (grp < 1)
          afb[i] = frag_a(wrow + (size_t)(mt + 5) * 16 * C_DIM + kt * 32, lane);
        acc[mt] = r;
      }
    }
    // pipeline choreography: preload 12 loads, then (1 WMMA, 2 loads) x5, 5 WMMAs
    SGB(0x020, 12);
#pragma unroll
    for (int i = 0; i < 5; ++i) { SGB(0x008, 1); SGB(0x020, 2); }
    SGB(0x008, 5);
  }

  const int s = sbase + l15;
  const int mo = (lane < 16) ? 0 : 8;
#pragma unroll
  for (int mt = 0; mt < 10; ++mt) {
#pragma unroll
    for (int j = 0; j < 8; ++j) {
      int m = (mbase + mt) * 16 + j + mo;
      __bf16 b = (__bf16)(acc[mt][j] + lds_bias[m]);
      if (m < 32)      Qt[((size_t)n * S_TOT + s) * MID + m] = b;
      else if (m < 64) Kt[((size_t)n * S_TOT + s) * MID + (m - 32)] = b;
      else             Vv[((size_t)n * C_DIM + (m - 64)) * S_TOT + s] = b;
    }
  }
}

// ---------------- kernel 4: fused flash-attention + residual.
// Each wave owns 16 query rows; block = 4 waves (64 s). Streams t in chunks of 128.
__global__ void __launch_bounds__(128)
attn_fused(const float* __restrict__ x, const __bf16* __restrict__ Qt,
           const __bf16* __restrict__ Kt, const __bf16* __restrict__ Vv,
           const float* __restrict__ gamma, float* __restrict__ out) {
  __shared__ __align__(16) __bf16 lds_P[4][16][128];   // 16 KB: probs, [s][t]
  __shared__ __align__(16) __bf16 lds_V[256][128];     // 64 KB: V chunk, [d][t]
  __shared__ float lds_scale[4][16];
  __shared__ float lds_l[4][16];

  const int tid = threadIdx.x;
  const int w = tid >> 5, lane = tid & 31, l15 = lane & 15;
  const int n = blockIdx.y;
  const int sbase = blockIdx.x * 64 + w * 16;

  const __bf16* Qn = Qt + (size_t)n * S_TOT * MID;
  const __bf16* Kn = Kt + (size_t)n * S_TOT * MID;
  const __bf16* Vn = Vv + (size_t)n * C_DIM * S_TOT;

  const v16bf qf = frag_a(Qn + (size_t)(sbase + l15) * MID, lane);
  const v8f zero = (v8f){0, 0, 0, 0, 0, 0, 0, 0};

  v8f O[16];
#pragma unroll
  for (int i = 0; i < 16; ++i) O[i] = zero;
  float m[8], l[8];
#pragma unroll
  for (int j = 0; j < 8; ++j) { m[j] = -3.0e38f; l[j] = 0.0f; }

#pragma unroll 1
  for (int ch = 0; ch < 32; ++ch) {
    const int tbase = ch * 128;
    __syncthreads();  // lds_V WAR: previous iteration's readers are done
    {  // stage V chunk [256 x 128] bf16 into LDS with async copies (ASYNCcnt)
      const unsigned long long vbase = (unsigned long long)(uintptr_t)(Vn + tbase);
#pragma unroll
      for (int i = 0; i < 32; ++i) {
        int linear = i * 128 + tid;            // 0..4095
        int d = linear >> 4;
        int c8 = (linear & 15) * 8;
        unsigned lds_off = (unsigned)(uintptr_t)(&lds_V[d][c8]);
        unsigned goff = (unsigned)((d << 13) + c8 * 2);   // d*S_TOT*2 + c8*2 bytes
        async_load_b128(lds_off, goff, vbase);
      }
    }
    // warm L2 for next chunk's K and V while we compute on this one
    if (ch < 31) {
      __builtin_prefetch(Kn + (size_t)(tbase + 128 + tid) * MID, 0, 0);
#pragma unroll
      for (int i = 0; i < 2; ++i) {
        int d = i * 128 + tid;
        __builtin_prefetch(Vn + (size_t)d * S_TOT + tbase + 128, 0, 0);
        __builtin_prefetch(Vn + (size_t)d * S_TOT + tbase + 192, 0, 0);
      }
    }
    // phase 1: scores D[s,t] = Q^T K (8 WMMAs), K frags pipelined 4 deep
    v8f D[8];
    {
      v16bf kfb[4];
#pragma unroll
      for (int i = 0; i < 4; ++i)
        kfb[i] = frag_b(Kn + (size_t)(tbase + i * 16 + l15) * MID, lane);
#pragma unroll
      for (int g = 0; g < 2; ++g) {
#pragma unroll
        for (int i = 0; i < 4; ++i) {
          const int tt = g * 4 + i;
          v8f r = __builtin_amdgcn_wmma_f32_16x16x32_bf16(false, qf, false, kfb[i],
                                                          (short)0, zero, false, false);
          if (g == 0)
            kfb[i] = frag_b(Kn + (size_t)(tbase + (tt + 4) * 16 + l15) * MID, lane);
          D[tt] = r;
        }
      }
      // choreography: preload 8 loads, then (1 WMMA, 2 loads) x4, 4 WMMAs
      SGB(0x020, 8);
#pragma unroll
      for (int i = 0; i < 4; ++i) { SGB(0x008, 1); SGB(0x020, 2); }
      SGB(0x008, 4);
    }
    // phase 2: online row max (lanes 0-15 hold rows j, lanes 16-31 rows j+8)
    float scj[8];
#pragma unroll
    for (int j = 0; j < 8; ++j) {
      float v = D[0][j];
#pragma unroll
      for (int tt = 1; tt < 8; ++tt) v = fmaxf(v, D[tt][j]);
      v = fmaxf(v, __shfl_xor(v, 1));
      v = fmaxf(v, __shfl_xor(v, 2));
      v = fmaxf(v, __shfl_xor(v, 4));
      v = fmaxf(v, __shfl_xor(v, 8));
      float mn = fmaxf(m[j], v);
      scj[j] = __expf(m[j] - mn);
      m[j] = mn;
      if (lane == j)      lds_scale[w][j] = scj[j];
      if (lane == j + 16) lds_scale[w][j + 8] = scj[j];
    }
    // phase 3: exp, write P (bf16) to LDS in [s][t] layout, row sums
    const int rrow = (lane < 16) ? 0 : 8;
#pragma unroll
    for (int j = 0; j < 8; ++j) {
      float rs = 0.0f;
#pragma unroll
      for (int tt = 0; tt < 8; ++tt) {
        float p = __expf(D[tt][j] - m[j]);
        rs += p;
        lds_P[w][j + rrow][tt * 16 + l15] = (__bf16)p;
      }
      rs += __shfl_xor(rs, 1);
      rs += __shfl_xor(rs, 2);
      rs += __shfl_xor(rs, 4);
      rs += __shfl_xor(rs, 8);
      l[j] = l[j] * scj[j] + rs;
    }
    // phase 4: rescale accumulator; O columns are s_local = lane&15
    float ssc = lds_scale[w][l15];
#pragma unroll
    for (int i = 0; i < 16; ++i) O[i] *= ssc;
    // make our async V copies visible, then block-wide barrier
    wait_asynccnt0();
    __syncthreads();
    // phase 5: O[d,s] += V[d,t] * P[s,t]^T  (64 WMMAs)
    // kk-outer/dt-inner: consecutive WMMAs hit different accumulators; V frags
    // pipelined 4 deep from LDS (8 DS loads in flight).
    v16bf pb[4];
#pragma unroll
    for (int kk = 0; kk < 4; ++kk) pb[kk] = frag_b(&lds_P[w][l15][kk * 32], lane);
    v16bf vfb[4];
#pragma unroll
    for (int i = 0; i < 4; ++i) vfb[i] = frag_a(&lds_V[i * 16 + l15][0], lane);
#pragma unroll
    for (int grp = 0; grp < 16; ++grp) {
      const int kk = grp >> 2;
      const int dt0 = (grp & 3) * 4;
#pragma unroll
      for (int i = 0; i < 4; ++i) {
        const int dt = dt0 + i;
        v8f r = __builtin_amdgcn_wmma_f32_16x16x32_bf16(false, vfb[i], false, pb[kk],
                                                        (short)0, O[dt], false, false);
        if (grp < 15) {
          const int ngrp = grp + 1;
          const int nkk = ngrp >> 2;
          const int ndt = (ngrp & 3) * 4 + i;
          vfb[i] = frag_a(&lds_V[ndt * 16 + l15][nkk * 32], lane);
        }
        O[dt] = r;
      }
    }
    // choreography: preload 16 DS reads (pb + vfb), then (1 WMMA, 2 DS) x60, 4 WMMAs
    SGB(0x100, 16);
#pragma unroll
    for (int i = 0; i < 60; ++i) { SGB(0x008, 1); SGB(0x100, 2); }
    SGB(0x008, 4);
  }
  // epilogue: divide by softmax denominator, residual + gamma, store
#pragma unroll
  for (int j = 0; j < 8; ++j) {
    if (lane == j)      lds_l[w][j] = l[j];
    if (lane == j + 16) lds_l[w][j + 8] = l[j];
  }
  const float rl = 1.0f / lds_l[w][l15];
  const float g = gamma[0];
  const int sidx = sbase + l15;
  const int dof = (lane < 16) ? 0 : 8;
#pragma unroll
  for (int dt = 0; dt < 16; ++dt) {
#pragma unroll
    for (int j = 0; j < 8; ++j) {
      int d = dt * 16 + j + dof;
      size_t idx = ((size_t)n * C_DIM + d) * S_TOT + sidx;
      out[idx] = x[idx] + g * O[dt][j] * rl;
    }
  }
}

extern "C" void kernel_launch(void* const* d_in, const int* in_sizes, int n_in,
                              void* d_out, int out_size, void* d_ws, size_t ws_size,
                              hipStream_t stream) {
  (void)in_sizes; (void)n_in; (void)out_size; (void)ws_size;
  const float* x     = (const float*)d_in[0];
  const float* Wq    = (const float*)d_in[1];
  const float* bq    = (const float*)d_in[2];
  const float* Wk    = (const float*)d_in[3];
  const float* bk    = (const float*)d_in[4];
  const float* Wv    = (const float*)d_in[5];
  const float* bv    = (const float*)d_in[6];
  const float* gamma = (const float*)d_in[7];
  float* out = (float*)d_out;

  char* ws = (char*)d_ws;
  __bf16* xbT  = (__bf16*)(ws);                // [N,S,C]  8 MB
  __bf16* Qt   = (__bf16*)(ws + 8388608);      // [N,S,32] 1 MB
  __bf16* Kt   = (__bf16*)(ws + 9437184);      // [N,S,32] 1 MB
  __bf16* Vv   = (__bf16*)(ws + 10485760);     // [N,C,S]  8 MB
  __bf16* Wall = (__bf16*)(ws + 18874368);     // [320,256] 160 KB
  float*  ball = (float*)(ws + 19038208);      // [320]     1.25 KB

  pack_weights<<<dim3(320), dim3(256), 0, stream>>>(Wq, bq, Wk, bk, Wv, bv, Wall, ball);
  transpose_x<<<dim3(S_TOT / 64, C_DIM / 64, NB), dim3(256), 0, stream>>>(x, xbT);
  qkv_gemm<<<dim3(S_TOT / 128, NB), dim3(512), 0, stream>>>(xbT, Wall, ball, Qt, Kt, Vv);
  attn_fused<<<dim3(S_TOT / 64, NB), dim3(128), 0, stream>>>(x, Qt, Kt, Vv, gamma, out);
}
